// SAE_23931557773437
// MI455X (gfx1250) — matrix operand order repaired and checked by reading, not compiled, not run on hardware
//
#include <hip/hip_runtime.h>
#include <hip/hip_bf16.h>

#define B_ROWS 8192
#define D_DIM  1024
#define K_DIM  32768
#define TOPK_N 32

#define BM 128
#define BN 128
#define RT 64
#define USE_ASYNC_LDS 1

typedef __attribute__((ext_vector_type(16))) __bf16 v16bf;
typedef __attribute__((ext_vector_type(8)))  __bf16 v8bf;
typedef __attribute__((ext_vector_type(8)))  float  v8f;

static __device__ __forceinline__ unsigned short f32_to_bf16(float f) {
  union { float f; unsigned u; } v; v.f = f;
  unsigned u = v.u;
  u += 0x7FFFu + ((u >> 16) & 1u);           // round-to-nearest-even
  return (unsigned short)(u >> 16);
}
static __device__ __forceinline__ float bf16_to_f32(unsigned short h) {
  union { unsigned u; float f; } v; v.u = ((unsigned)h) << 16;
  return v.f;
}

// ---------------- Kernel 1: Xbf = bf16(X - b) --------------------------------
__global__ void __launch_bounds__(256) k_prep_x(const float* __restrict__ X,
                                                const float* __restrict__ bvec,
                                                unsigned short* __restrict__ Xbf) {
  int id = blockIdx.x * 256 + threadIdx.x;         // one float4 per thread
  float4 x4 = ((const float4*)X)[id];
  int d = (id * 4) & (D_DIM - 1);
  float4 b4 = *(const float4*)(bvec + d);
  ushort4 o;
  o.x = f32_to_bf16(x4.x - b4.x);
  o.y = f32_to_bf16(x4.y - b4.y);
  o.z = f32_to_bf16(x4.z - b4.z);
  o.w = f32_to_bf16(x4.w - b4.w);
  ((ushort4*)Xbf)[id] = o;
}

// ---------------- Kernel 2: WbfT[k][d] = bf16(W[d][k]) (tiled transpose) -----
__global__ void __launch_bounds__(256) k_prep_wt(const float* __restrict__ W,
                                                 unsigned short* __restrict__ WbfT) {
  __shared__ float tile[32][33];
  int k0 = blockIdx.x * 32;
  int d0 = blockIdx.y * 32;
  int t  = threadIdx.x;
  int r  = t >> 3;            // 0..31
  int c4 = (t & 7) * 4;       // 0,4,...,28
  float4 w4 = *(const float4*)(W + (size_t)(d0 + r) * K_DIM + k0 + c4);
  tile[r][c4 + 0] = w4.x; tile[r][c4 + 1] = w4.y;
  tile[r][c4 + 2] = w4.z; tile[r][c4 + 3] = w4.w;
  __syncthreads();
  ushort4 o;                  // r is now the k index, c4 the d index
  o.x = f32_to_bf16(tile[c4 + 0][r]);
  o.y = f32_to_bf16(tile[c4 + 1][r]);
  o.z = f32_to_bf16(tile[c4 + 2][r]);
  o.w = f32_to_bf16(tile[c4 + 3][r]);
  *(ushort4*)(WbfT + (size_t)(k0 + r) * D_DIM + d0 + c4) = o;
}

// ---------------- Kernel 3: encode GEMM, C = ReLU(Xbf @ WbfT^T) --------------
// Workgroup tile 128x128, 8 waves (2x4), each wave 64x32 = 4x2 WMMA tiles.
// K-step 64 (two WMMA k-halves per LDS round), double-buffered async fills.
__global__ void __launch_bounds__(256) k_encode(const unsigned short* __restrict__ Xbf,
                                                const unsigned short* __restrict__ WbfT,
                                                float* __restrict__ C) {
  __shared__ __bf16 lx[2][BM][RT];   // 2*128*64*2 = 32 KiB
  __shared__ __bf16 lw[2][BN][RT];   // 32 KiB

  const int t    = threadIdx.x;
  const int lane = t & 31;
  const int w    = t >> 5;
  const int wm   = w >> 2;      // 0..1
  const int wn   = w & 3;       // 0..3
  const int bn0  = blockIdx.x * BN;
  const int bm0  = blockIdx.y * BM;

  // tile-fill assignment: thread -> 64B of one row (four b128 chunks)
  const int frow = t >> 1;              // 0..127
  const int fcol = (t & 1) * 32;        // bf16 element offset (0 or 32)
  const size_t gx = (size_t)(bm0 + frow) * D_DIM + fcol;
  const size_t gw = (size_t)(bn0 + frow) * D_DIM + fcol;

  auto fill = [&](int buf, int r0) {
    const unsigned short* sx = Xbf  + gx + r0;
    const unsigned short* sw = WbfT + gw + r0;
#if USE_ASYNC_LDS
    unsigned dx = (unsigned)(unsigned long long)&lx[buf][frow][fcol];
    unsigned dw = (unsigned)(unsigned long long)&lw[buf][frow][fcol];
    // IOFFSET is added to both the global and LDS address (ISA 15.18.3)
    asm volatile("global_load_async_to_lds_b128 %0, %1, off"            :: "v"(dx), "v"(sx) : "memory");
    asm volatile("global_load_async_to_lds_b128 %0, %1, off offset:16" :: "v"(dx), "v"(sx) : "memory");
    asm volatile("global_load_async_to_lds_b128 %0, %1, off offset:32" :: "v"(dx), "v"(sx) : "memory");
    asm volatile("global_load_async_to_lds_b128 %0, %1, off offset:48" :: "v"(dx), "v"(sx) : "memory");
    asm volatile("global_load_async_to_lds_b128 %0, %1, off"            :: "v"(dw), "v"(sw) : "memory");
    asm volatile("global_load_async_to_lds_b128 %0, %1, off offset:16" :: "v"(dw), "v"(sw) : "memory");
    asm volatile("global_load_async_to_lds_b128 %0, %1, off offset:32" :: "v"(dw), "v"(sw) : "memory");
    asm volatile("global_load_async_to_lds_b128 %0, %1, off offset:48" :: "v"(dw), "v"(sw) : "memory");
#else
    #pragma unroll
    for (int q = 0; q < 4; ++q) {
      uint4 a = ((const uint4*)sx)[q];
      uint4 b = ((const uint4*)sw)[q];
      ((uint4*)&lx[buf][frow][fcol])[q] = a;
      ((uint4*)&lw[buf][frow][fcol])[q] = b;
    }
#endif
  };

  v8f zero = {0.f,0.f,0.f,0.f,0.f,0.f,0.f,0.f};
  v8f acc[4][2];
  #pragma unroll
  for (int i = 0; i < 4; ++i)
    #pragma unroll
    for (int j = 0; j < 2; ++j) acc[i][j] = zero;

  fill(0, 0);
  const int mrow = lane & 15;
  const int kb   = (lane >> 4) * 8;       // A frag: K pairs 0..7 / 8..15 halves
  const int ncol = lane & 15;
  const int rb   = (lane >> 4) * 8;       // B frag: K pairs 0..7 / 8..15 halves

  for (int it = 0; it < D_DIM / RT; ++it) {
    const int cur = it & 1;
#if USE_ASYNC_LDS
    asm volatile("s_wait_asynccnt 0" ::: "memory");
#endif
    __syncthreads();
    if (it + 1 < D_DIM / RT) fill(cur ^ 1, (it + 1) * RT);

    #pragma unroll
    for (int kh = 0; kh < 2; ++kh) {
      const int ko = kh * 32;
      v16bf a[4], bb[2];
      #pragma unroll
      for (int i = 0; i < 4; ++i) {
        v8bf lo = *(const v8bf*)&lx[cur][wm * 64 + i * 16 + mrow][ko + kb];
        v8bf hi = *(const v8bf*)&lx[cur][wm * 64 + i * 16 + mrow][ko + kb + 16];
        a[i] = __builtin_shufflevector(lo, hi, 0,1,2,3,4,5,6,7,8,9,10,11,12,13,14,15);
      }
      #pragma unroll
      for (int j = 0; j < 2; ++j) {
        v8bf lo = *(const v8bf*)&lw[cur][wn * 32 + j * 16 + ncol][ko + rb];
        v8bf hi = *(const v8bf*)&lw[cur][wn * 32 + j * 16 + ncol][ko + rb + 16];
        bb[j] = __builtin_shufflevector(lo, hi, 0,1,2,3,4,5,6,7,8,9,10,11,12,13,14,15);
      }
      #pragma unroll
      for (int i = 0; i < 4; ++i)
        #pragma unroll
        for (int j = 0; j < 2; ++j)
          acc[i][j] = __builtin_amdgcn_wmma_f32_16x16x32_bf16(
              false, a[i], false, bb[j], (short)0, acc[i][j], false, false);
    }
  }

  // Epilogue: ReLU + store. C/D layout: VGPR v -> M = v + 8*(lane>>4), N = lane&15
  #pragma unroll
  for (int i = 0; i < 4; ++i) {
    int grow = bm0 + wm * 64 + i * 16 + 8 * (lane >> 4);
    #pragma unroll
    for (int j = 0; j < 2; ++j) {
      int gcol = bn0 + wn * 32 + j * 16 + (lane & 15);
      #pragma unroll
      for (int v = 0; v < 8; ++v) {
        float x = fmaxf(acc[i][j][v], 0.0f);
        C[(size_t)(grow + v) * K_DIM + gcol] = x;
      }
    }
  }
}

// ---------------- Kernel 4: exact top-32 radix-select + mask + fused decode --
__global__ void __launch_bounds__(512) k_topk_decode(float* __restrict__ C,
                                                     float* __restrict__ Xh,
                                                     const unsigned short* __restrict__ WbfT,
                                                     const float* __restrict__ bvec) {
  extern __shared__ unsigned smem[];
  unsigned* hist           = smem;                 // 2048
  volatile unsigned* scal  = smem + 2048;          // 16 scalars
  unsigned* eq             = smem + 2064;          // 256 tie indices
  unsigned* lk             = smem + 2320;          // 32 kept indices
  float*    lc             = (float*)(smem + 2352);// 32 kept values
  unsigned* gsum           = smem + 2384;          // 32 group sums

  const int t   = threadIdx.x;
  const int row = blockIdx.x;
  float4* Crow = (float4*)(C + (size_t)row * K_DIM);

  float4 r[16];
  #pragma unroll
  for (int i = 0; i < 16; ++i) r[i] = Crow[i * 512 + t];   // coalesced

  auto for_elems = [&](auto&& fn) {
    #pragma unroll
    for (int i = 0; i < 16; ++i) {
      int base = (i * 512 + t) * 4;
      fn(r[i].x, base + 0); fn(r[i].y, base + 1);
      fn(r[i].z, base + 2); fn(r[i].w, base + 3);
    }
  };

  // Two-level suffix scan: lanes 0..31 each sum a chunk, thread 0 walks
  // <=32 group sums + <=chunk bins instead of nbins serial steps.
  // Writes: scal[ob] = crossing bin, scal[og] = count strictly above it.
  auto find_threshold = [&](int nbins, unsigned need, unsigned prev_greater,
                            int ob, int og) {
    const int chunk = nbins >> 5;
    if (t < 32) {
      unsigned s = 0;
      for (int q = 0; q < chunk; ++q) s += hist[t * chunk + q];
      gsum[t] = s;
    }
    __syncthreads();
    if (t == 0) {
      unsigned cum = 0; int grp = 31;
      while (grp > 0 && cum + gsum[grp] < need) { cum += gsum[grp]; --grp; }
      int b = grp * chunk + chunk - 1;
      while (b > grp * chunk && cum + hist[b] < need) { cum += hist[b]; --b; }
      scal[ob] = (unsigned)b;
      scal[og] = prev_greater + cum;     // strictly greater than bin b
    }
    __syncthreads();
  };

  // ---- radix pass 1: bits [31:21] ----
  for (int j = t; j < 2048; j += 512) hist[j] = 0;
  __syncthreads();
  for_elems([&](float& fv, int) { atomicAdd(&hist[__float_as_uint(fv) >> 21], 1u); });
  __syncthreads();
  find_threshold(2048, TOPK_N, 0, 0, 1);
  unsigned b1 = scal[0], g1 = scal[1];

  // ---- radix pass 2: bits [20:10] ----
  for (int j = t; j < 2048; j += 512) hist[j] = 0;
  __syncthreads();
  for_elems([&](float& fv, int) {
    unsigned u = __float_as_uint(fv);
    if ((u >> 21) == b1) atomicAdd(&hist[(u >> 10) & 0x7FFu], 1u);
  });
  __syncthreads();
  find_threshold(2048, TOPK_N - g1, g1, 2, 3);
  unsigned b2 = scal[2], g2 = scal[3];
  unsigned pre = (b1 << 11) | b2;

  // ---- radix pass 3: bits [9:0] ----
  for (int j = t; j < 1024; j += 512) hist[j] = 0;
  __syncthreads();
  for_elems([&](float& fv, int) {
    unsigned u = __float_as_uint(fv);
    if ((u >> 10) == pre) atomicAdd(&hist[u & 0x3FFu], 1u);
  });
  __syncthreads();
  find_threshold(1024, TOPK_N - g2, g2, 4, 5);
  if (t == 0) {
    scal[4] = (b1 << 21) | (b2 << 10) | scal[4];       // full threshold bits
    scal[6] = 0; scal[7] = 0;                          // eqcnt, listcnt
  }
  __syncthreads();
  unsigned T = scal[4], cg = scal[5], rem = TOPK_N - cg;

  // ---- tie-break (first `rem` equal values by index, matching lax.top_k) ----
  if (T != 0) {
    for_elems([&](float& fv, int ei) {
      if (__float_as_uint(fv) == T) {
        unsigned p = atomicAdd((unsigned*)&scal[6], 1u);
        if (p < 256) eq[p] = (unsigned)ei;
      }
    });
    __syncthreads();
    if (t == 0) {
      int n = scal[6] < 256u ? (int)scal[6] : 256;
      for (int a = 1; a < n; ++a) {                    // tiny insertion sort
        unsigned key = eq[a]; int c = a - 1;
        while (c >= 0 && eq[c] > key) { eq[c + 1] = eq[c]; --c; }
        eq[c + 1] = key;
      }
      scal[8] = (unsigned)n;
    }
    __syncthreads();
  }
  unsigned neq = (T != 0) ? scal[8] : 0;
  unsigned nkeep_eq = rem < neq ? rem : neq;

  // ---- apply mask, collect (k, c) list ----
  for_elems([&](float& fv, int ei) {
    unsigned u = __float_as_uint(fv);
    bool keep;
    if (T == 0)        keep = (u > 0);
    else if (u > T)    keep = true;
    else if (u == T) {
      keep = false;
      for (unsigned q = 0; q < nkeep_eq; ++q)
        if (eq[q] == (unsigned)ei) { keep = true; break; }
    } else keep = false;
    if (!keep) fv = 0.0f;
    else if (fv > 0.0f) {
      unsigned p = atomicAdd((unsigned*)&scal[7], 1u);
      if (p < TOPK_N) { lk[p] = (unsigned)ei; lc[p] = fv; }
    }
  });
  __syncthreads();
  if (t == 0) {                        // sort list by index -> deterministic sum
    int n = scal[7] < (unsigned)TOPK_N ? (int)scal[7] : TOPK_N;
    for (int a = 1; a < n; ++a) {
      unsigned key = lk[a]; float kc = lc[a]; int c = a - 1;
      while (c >= 0 && lk[c] > key) { lk[c + 1] = lk[c]; lc[c + 1] = lc[c]; --c; }
      lk[c + 1] = key; lc[c + 1] = kc;
    }
    scal[9] = (unsigned)n;
  }
  __syncthreads();

  // ---- write back masked C ----
  #pragma unroll
  for (int i = 0; i < 16; ++i) Crow[i * 512 + t] = r[i];

  // ---- fused decode: Xh[row] = sum_l c_l * WbfT[k_l][:] + b ----
  int nl = (int)scal[9];
  int d0 = t * 2;
  float a0 = 0.f, a1 = 0.f;
  for (int l = 0; l < nl; ++l) {
    unsigned k = lk[l]; float c = lc[l];
    unsigned w2 = *(const unsigned*)(WbfT + (size_t)k * D_DIM + d0);
    a0 += c * bf16_to_f32((unsigned short)(w2 & 0xFFFFu));
    a1 += c * bf16_to_f32((unsigned short)(w2 >> 16));
  }
  float2 o; o.x = a0 + bvec[d0]; o.y = a1 + bvec[d0 + 1];
  *(float2*)(Xh + (size_t)row * D_DIM + d0) = o;
}

// -----------------------------------------------------------------------------
extern "C" void kernel_launch(void* const* d_in, const int* in_sizes, int n_in,
                              void* d_out, int out_size, void* d_ws, size_t ws_size,
                              hipStream_t stream) {
  const float* X  = (const float*)d_in[0];
  const float* W  = (const float*)d_in[1];
  const float* bv = (const float*)d_in[2];
  float* Xh = (float*)d_out;
  float* C  = (float*)d_out + (size_t)B_ROWS * D_DIM;

  unsigned short* Xbf  = (unsigned short*)d_ws;                         // 16 MiB
  unsigned short* WbfT = (unsigned short*)((char*)d_ws + (1ull << 24)); // 64 MiB

  k_prep_x<<<dim3((B_ROWS * D_DIM / 4) / 256), 256, 0, stream>>>(X, bv, Xbf);
  k_prep_wt<<<dim3(K_DIM / 32, D_DIM / 32), 256, 0, stream>>>(W, WbfT);
  k_encode<<<dim3(K_DIM / BN, B_ROWS / BM), 256, 0, stream>>>(Xbf, WbfT, C);
  size_t smem = 2416 * sizeof(unsigned);
  k_topk_decode<<<dim3(B_ROWS), 512, smem, stream>>>(C, Xh, WbfT, bv);
}
